// CasualAttention_70342974374206
// MI455X (gfx1250) — compile-verified, matrix operand order checked by reading, not run on hardware
//
#include <hip/hip_runtime.h>

// Causal attention (B=4, S=2048, D=1024) for MI455X / gfx1250.
//
// Roofline: ~120 GFLOP vs ~80 MB real I/O -> matrix-core bound; everything
// runs through v_wmma_f32_16x16x32_bf16 (f32 accumulate).
//
// Data movement (CDNA5 path): each 256-thread block (8 waves, 2x4 grid of
// 64x64 wave tiles -> 128x256 block tile) stages its A/B panels for each
// K-step of 32 into LDS with global_load_async_to_lds_b128 (ASYNCcnt),
// double-buffered so the async copy of step t+1 overlaps the 16 WMMAs/wave
// of step t. Unique panel data per K-step is 24KB vs 64KB if each wave
// loaded its own fragments -> 2.7x cut in WGP$/L2 traffic. Fragments are
// read from LDS with ds_load_b128; panel rows are padded 64B->80B so the
// 16-row fragment reads map to distinct banks (20*l mod 64 is a permutation).
//
// Causal structure: score blocks entirely above the diagonal skip their
// K-loop (block-uniform), and weights@v clamps K to the nonzero band.

#define B_DIM 4
#define S_DIM 2048
#define D_DIM 1024
#define BS_DIM (B_DIM * S_DIM)          // 8192 tokens
#define INV_SQRT_D 0.03125f             // 1/sqrt(1024)

#define ROW_BYTES 80                    // 32 bf16 = 64B, padded to 80B
#define A_PANEL_BYTES (128 * ROW_BYTES) // 10240
#define B_PANEL_BYTES (256 * ROW_BYTES) // 20480
#define BUF_BYTES (A_PANEL_BYTES + B_PANEL_BYTES)   // 30720
#define SMEM_BYTES (2 * BUF_BYTES)                  // 61440 (<64KB static)

typedef __attribute__((ext_vector_type(16))) __bf16 v16bf;
typedef __attribute__((ext_vector_type(8)))  float  v8f;

struct bf16frag_bits { uint4 lo, hi; };

__device__ __forceinline__ __bf16 f32_to_bf16(float f) {
  unsigned u = __float_as_uint(f);
  u += 0x7FFFu + ((u >> 16) & 1u);      // round-to-nearest-even
  unsigned short h = (unsigned short)(u >> 16);
  return __builtin_bit_cast(__bf16, h);
}

__device__ __forceinline__ v8f wmma_bf16(v16bf a, v16bf b, v8f c) {
  return __builtin_amdgcn_wmma_f32_16x16x32_bf16(
      /*neg_a=*/false, a, /*neg_b=*/false, b,
      /*c_mod=*/(short)0, c, /*reuse_a=*/false, /*reuse_b=*/false);
}

// ---- CDNA5 async global->LDS copy (ASYNCcnt-tracked, no VGPR round trip) ----
// LDS destination address is taken from VDST; low 32 bits of a generic LDS pointer are the
// wave-relative LDS byte address (ISA 10.2 aperture rules).
__device__ __forceinline__ void async_cp16(void* lds, const void* gsrc) {
  unsigned laddr = (unsigned)(uintptr_t)lds;
  asm volatile("global_load_async_to_lds_b128 %0, %1, off"
               :: "v"(laddr), "v"(gsrc) : "memory");
}
__device__ __forceinline__ void wait_async0() {
  asm volatile("s_wait_asynccnt 0" ::: "memory");
}

// Fragment layout (ISA 7.12.2), served from an LDS panel whose row r holds
// K=0..31 of panel row r at byte offset r*ROW_BYTES: lane<16 -> row=lane,
// byte chunks {0,32}; lane>=16 -> row=lane-16, byte chunks {16,48}.
__device__ __forceinline__ v16bf lds_frag(const char* panel, int row0) {
  const int lane = threadIdx.x & 31;
  const int row  = row0 + (lane & 15);
  const int boff = (lane >> 4) << 4;    // 0 or 16 bytes (K-half select)
  const char* p = panel + row * ROW_BYTES + boff;
  bf16frag_bits t;
  t.lo = *reinterpret_cast<const uint4*>(p);        // K = koff .. koff+7
  t.hi = *reinterpret_cast<const uint4*>(p + 32);   // K = 16+koff .. +7
  return __builtin_bit_cast(v16bf, t);
}

// Cooperative async fill of one K-step's panels: A = 128 rows (2 chunks/thr),
// B = 256 rows (4 chunks/thr); each chunk is one b128.
__device__ __forceinline__ void fill_panels(const __bf16* __restrict__ A,
                                            const __bf16* __restrict__ Bt,
                                            int lda, int ldb, int k0,
                                            char* buf) {
  const int tid = threadIdx.x;
  const __bf16* Ak = A + k0;
  const __bf16* Bk = Bt + k0;
#pragma unroll
  for (int t = 0; t < 2; ++t) {         // A panel: 512 chunks
    int q = tid + t * 256;
    int row = q >> 2, c = (q & 3) << 4; // c in bytes
    async_cp16(buf + row * ROW_BYTES + c,
               (const char*)(Ak + (size_t)row * lda) + c);
  }
#pragma unroll
  for (int t = 0; t < 4; ++t) {         // B panel: 1024 chunks
    int q = tid + t * 256;
    int row = q >> 2, c = (q & 3) << 4;
    async_cp16(buf + A_PANEL_BYTES + row * ROW_BYTES + c,
               (const char*)(Bk + (size_t)row * ldb) + c);
  }
}

// ---------------- LDS-staged, double-buffered 128x256 block GEMM core ------
// A: block panel origin (128 rows, ld=lda); Bt: block panel origin (256 rows,
// ld=ldb); kend multiple of 32, block-uniform. mw/nw: wave tile origin inside
// the block panels. acc: 4x4 grid of 16x16 f32 tiles.
__device__ __forceinline__ void gemm_core_block(const __bf16* __restrict__ A,
                                                const __bf16* __restrict__ Bt,
                                                int lda, int ldb, int kend,
                                                int mw, int nw,
                                                v8f acc[4][4], char* smem) {
  const int ktiles = kend >> 5;
  fill_panels(A, Bt, lda, ldb, 0, smem);
#pragma unroll 1
  for (int kt = 0; kt < ktiles; ++kt) {
    char* cur = smem + (kt & 1) * BUF_BYTES;
    wait_async0();                      // own wave's fill of `cur` done
    __syncthreads();                    // everyone's fill of `cur` done
    if (kt + 1 < ktiles)                // overlap next copy with compute
      fill_panels(A, Bt, lda, ldb, (kt + 1) << 5,
                  smem + ((kt + 1) & 1) * BUF_BYTES);
    const char* ap = cur;
    const char* bp = cur + A_PANEL_BYTES;
    v16bf a[4], b[4];
#pragma unroll
    for (int i = 0; i < 4; ++i) a[i] = lds_frag(ap, mw + 16 * i);
#pragma unroll
    for (int j = 0; j < 4; ++j) b[j] = lds_frag(bp, nw + 16 * j);
#pragma unroll
    for (int i = 0; i < 4; ++i)
#pragma unroll
      for (int j = 0; j < 4; ++j)
        acc[i][j] = wmma_bf16(a[i], b[j], acc[i][j]);
    __syncthreads();                    // readers done before `cur` is refilled
  }
}

// C/D layout (ISA 7.12.2): VGPR r holds M = r + 8*(lane>=16), N = lane&15.
__device__ __forceinline__ void store_tile_bf16(__bf16* __restrict__ dst, size_t ld,
                                                int m0, int n0, v8f c) {
  const int lane = threadIdx.x & 31;
  const int n  = n0 + (lane & 15);
  const int mb = m0 + ((lane >> 4) << 3);
#pragma unroll
  for (int r = 0; r < 8; ++r)
    dst[(size_t)(mb + r) * ld + n] = f32_to_bf16(c[r]);
}

__device__ __forceinline__ void store_tile_bf16_T(__bf16* __restrict__ dst, size_t ld,
                                                  int m0, int n0, v8f c) {
  const int lane = threadIdx.x & 31;
  const int n  = n0 + (lane & 15);
  const int mb = m0 + ((lane >> 4) << 3);
#pragma unroll
  for (int r = 0; r < 8; ++r)
    dst[(size_t)n * ld + (mb + r)] = f32_to_bf16(c[r]);
}

__device__ __forceinline__ void store_tile_f32(float* __restrict__ dst, size_t ld,
                                               int m0, int n0, v8f c) {
  const int lane = threadIdx.x & 31;
  const int n  = n0 + (lane & 15);
  const int mb = m0 + ((lane >> 4) << 3);
#pragma unroll
  for (int r = 0; r < 8; ++r)
    dst[(size_t)(mb + r) * ld + n] = c[r];
}

__device__ __forceinline__ void store_scores_tile(float* __restrict__ dst,
                                                  int m0, int n0, v8f c) {
  const int lane = threadIdx.x & 31;
  const int n  = n0 + (lane & 15);
  const int mb = m0 + ((lane >> 4) << 3);
#pragma unroll
  for (int r = 0; r < 8; ++r) {
    const int i = mb + r;
    dst[(size_t)i * S_DIM + n] = (n > i) ? -__builtin_inff()
                                         : c[r] * INV_SQRT_D;
  }
}

// ---------------- converters ----------------

__global__ void __launch_bounds__(256)
cvt_f32_bf16_kernel(const float* __restrict__ in, __bf16* __restrict__ out, int n) {
  int i = blockIdx.x * 256 + threadIdx.x;
  if (i < n) out[i] = f32_to_bf16(in[i]);
}

// W [D_in, D_out] f32 -> Wt [D_out, D_in] bf16 (so B operand is K-contiguous)
__global__ void __launch_bounds__(256)
cvt_transpose_w_kernel(const float* __restrict__ in, __bf16* __restrict__ out) {
  int i = blockIdx.x * 256 + threadIdx.x;   // i = d * D + e, coalesced read
  int d = i >> 10;
  int e = i & (D_DIM - 1);
  out[(size_t)e * D_DIM + d] = f32_to_bf16(in[i]);
}

// ---------------- stage 1: q/k/v projections ----------------
__global__ void __launch_bounds__(256)
qkv_gemm_kernel(const __bf16* __restrict__ X, const __bf16* __restrict__ Wt,
                __bf16* __restrict__ Out, int transC) {
  __shared__ __align__(16) char smem[SMEM_BYTES];
  const int wave = threadIdx.x >> 5;
  const int mw = (wave >> 2) * 64;          // 0 or 64 in block A panel
  const int nw = (wave & 3) * 64;           // 0..192 in block B panel
  const int M0 = blockIdx.y * 128 + mw;
  const int N0 = blockIdx.x * 256 + nw;
  v8f acc[4][4] = {};
  gemm_core_block(X  + (size_t)(blockIdx.y * 128) * D_DIM,
                  Wt + (size_t)(blockIdx.x * 256) * D_DIM,
                  D_DIM, D_DIM, D_DIM, mw, nw, acc, smem);
  if (!transC) {
#pragma unroll
    for (int i = 0; i < 4; ++i)
#pragma unroll
      for (int j = 0; j < 4; ++j)
        store_tile_bf16(Out, D_DIM, M0 + 16 * i, N0 + 16 * j, acc[i][j]);
  } else {           // v stored transposed: vt[e, token], ld = BS
#pragma unroll
    for (int i = 0; i < 4; ++i)
#pragma unroll
      for (int j = 0; j < 4; ++j)
        store_tile_bf16_T(Out, BS_DIM, M0 + 16 * i, N0 + 16 * j, acc[i][j]);
  }
}

// ---------------- stage 2: scores = (q k^T) * scale, causal-masked ---------
// B operand is k row-major directly (Bt[n=j, k=e] = k[j, e]).
__global__ void __launch_bounds__(256)
scores_gemm_kernel(const __bf16* __restrict__ Q, const __bf16* __restrict__ Kb,
                   float* __restrict__ Sc) {
  __shared__ __align__(16) char smem[SMEM_BYTES];
  const int b = blockIdx.z;
  const __bf16* Qb = Q  + (size_t)b * S_DIM * D_DIM;
  const __bf16* Kt = Kb + (size_t)b * S_DIM * D_DIM;
  float* Sb = Sc + (size_t)b * S_DIM * S_DIM;
  const int wave = threadIdx.x >> 5;
  const int mw = (wave >> 2) * 64;
  const int nw = (wave & 3) * 64;
  const int M0 = blockIdx.y * 128 + mw;     // query i
  const int N0 = blockIdx.x * 256 + nw;     // key j
  v8f acc[4][4] = {};
  // Block-uniform causal skip (keeps barriers uniform); partially-masked
  // blocks compute everything and rely on the masked store.
  if ((int)(blockIdx.x * 256) <= (int)(blockIdx.y * 128) + 127)
    gemm_core_block(Qb + (size_t)(blockIdx.y * 128) * D_DIM,
                    Kt + (size_t)(blockIdx.x * 256) * D_DIM,
                    D_DIM, D_DIM, D_DIM, mw, nw, acc, smem);
#pragma unroll
  for (int i = 0; i < 4; ++i)
#pragma unroll
    for (int j = 0; j < 4; ++j)
      store_scores_tile(Sb, M0 + 16 * i, N0 + 16 * j, acc[i][j]);
}

// ---------------- stage 3: softmax over the QUERY axis (per key column j) --
__global__ void __launch_bounds__(256)
col_softmax_kernel(const float* __restrict__ Sc, __bf16* __restrict__ Wg) {
  const int lane = threadIdx.x & 31;
  const int wave = threadIdx.x >> 5;
  const int col  = blockIdx.x * 8 + wave;   // 0 .. B*S-1
  const int b = col >> 11;                  // / S_DIM
  const int j = col & (S_DIM - 1);
  const float* sc = Sc + (size_t)b * S_DIM * S_DIM + j;

  float m = -__builtin_inff();
  float sum = 0.0f;
  for (int i = j + lane; i < S_DIM; i += 32) {    // only finite entries
    float v = sc[(size_t)i * S_DIM];
    float nm = fmaxf(m, v);
    sum = sum * __expf(m - nm) + __expf(v - nm);  // m=-inf only on first iter
    m = nm;
  }
#pragma unroll
  for (int off = 16; off > 0; off >>= 1) {
    float om = __shfl_xor(m, off, 32);
    float os = __shfl_xor(sum, off, 32);
    float nm = fmaxf(m, om);
    float sa = (m  == nm) ? 1.0f : __expf(m  - nm);   // guards -inf == -inf
    float sb = (om == nm) ? 1.0f : __expf(om - nm);
    sum = sum * sa + os * sb;
    m = nm;
  }
  const float inv = 1.0f / sum;
  __bf16* wp = Wg + (size_t)b * S_DIM * S_DIM + j;
  for (int i = lane; i < S_DIM; i += 32) {
    float w = 0.0f;
    if (i >= j) w = __expf(sc[(size_t)i * S_DIM] - m) * inv;
    wp[(size_t)i * S_DIM] = f32_to_bf16(w);
  }
}

// ---------------- stage 4: out = weights @ v ----------------
// A = weights [S,S] bf16; B operand = vt [D, BS] bf16. Block-uniform K clamp:
// weights[i, j] == 0 for j > i, so K beyond blockM+128 multiplies exact zeros.
__global__ void __launch_bounds__(256)
out_gemm_kernel(const __bf16* __restrict__ Wg, const __bf16* __restrict__ Vt,
                float* __restrict__ Out) {
  __shared__ __align__(16) char smem[SMEM_BYTES];
  const int b = blockIdx.z;
  const __bf16* Ab = Wg + (size_t)b * S_DIM * S_DIM;
  const int wave = threadIdx.x >> 5;
  const int mw = (wave >> 2) * 64;
  const int nw = (wave & 3) * 64;
  const int M0 = blockIdx.y * 128 + mw;     // token i
  const int N0 = blockIdx.x * 256 + nw;     // feature e
  const int kblk = blockIdx.y * 128 + 128;  // block-uniform causal band
  const int kend = (kblk < S_DIM) ? kblk : S_DIM;
  v8f acc[4][4] = {};
  gemm_core_block(Ab + (size_t)(blockIdx.y * 128) * S_DIM,
                  Vt + (size_t)(blockIdx.x * 256) * BS_DIM + (size_t)b * S_DIM,
                  S_DIM, BS_DIM, kend, mw, nw, acc, smem);
  float* Ob = Out + (size_t)b * S_DIM * D_DIM;
#pragma unroll
  for (int i = 0; i < 4; ++i)
#pragma unroll
    for (int j = 0; j < 4; ++j)
      store_tile_f32(Ob, D_DIM, M0 + 16 * i, N0 + 16 * j, acc[i][j]);
}

// ---------------- host launch ----------------

extern "C" void kernel_launch(void* const* d_in, const int* in_sizes, int n_in,
                              void* d_out, int out_size, void* d_ws, size_t ws_size,
                              hipStream_t stream) {
  (void)in_sizes; (void)n_in; (void)out_size; (void)ws_size;
  const float* x  = (const float*)d_in[0];
  const float* Wq = (const float*)d_in[1];
  const float* Wk = (const float*)d_in[2];
  const float* Wv = (const float*)d_in[3];
  float* out = (float*)d_out;

  // Workspace carve-out (~174 MB total, all sizes 256B-aligned):
  char* ws = (char*)d_ws;
  size_t off = 0;
  auto carve = [&](size_t bytes) -> char* {
    char* p = ws + off;
    off = (off + bytes + 255) & ~(size_t)255;
    return p;
  };
  __bf16* xb  = (__bf16*)carve((size_t)BS_DIM * D_DIM * 2);   // x bf16
  __bf16* wqt = (__bf16*)carve((size_t)D_DIM * D_DIM * 2);    // Wq^T bf16
  __bf16* wkt = (__bf16*)carve((size_t)D_DIM * D_DIM * 2);
  __bf16* wvt = (__bf16*)carve((size_t)D_DIM * D_DIM * 2);
  __bf16* qb  = (__bf16*)carve((size_t)BS_DIM * D_DIM * 2);   // q bf16 [BS, D]
  __bf16* kb  = (__bf16*)carve((size_t)BS_DIM * D_DIM * 2);   // k bf16 [BS, D]
  __bf16* vtb = (__bf16*)carve((size_t)BS_DIM * D_DIM * 2);   // v^T bf16 [D, BS]
  float*  sc  = (float*)carve((size_t)B_DIM * S_DIM * S_DIM * 4); // scores f32
  __bf16* wg  = (__bf16*)carve((size_t)B_DIM * S_DIM * S_DIM * 2); // weights bf16

  dim3 thr(256);
  cvt_f32_bf16_kernel<<<dim3((BS_DIM * D_DIM) / 256), thr, 0, stream>>>(
      x, xb, BS_DIM * D_DIM);
  cvt_transpose_w_kernel<<<dim3((D_DIM * D_DIM) / 256), thr, 0, stream>>>(Wq, wqt);
  cvt_transpose_w_kernel<<<dim3((D_DIM * D_DIM) / 256), thr, 0, stream>>>(Wk, wkt);
  cvt_transpose_w_kernel<<<dim3((D_DIM * D_DIM) / 256), thr, 0, stream>>>(Wv, wvt);

  dim3 gq(D_DIM / 256, BS_DIM / 128, 1);                // (4, 64)
  qkv_gemm_kernel<<<gq, thr, 0, stream>>>(xb, wqt, qb, 0);
  qkv_gemm_kernel<<<gq, thr, 0, stream>>>(xb, wkt, kb, 0);
  qkv_gemm_kernel<<<gq, thr, 0, stream>>>(xb, wvt, vtb, 1);

  dim3 gs(S_DIM / 256, S_DIM / 128, B_DIM);             // (8, 16, 4)
  scores_gemm_kernel<<<gs, thr, 0, stream>>>(qb, kb, sc);

  col_softmax_kernel<<<dim3(BS_DIM / 8), thr, 0, stream>>>(sc, wg);

  dim3 go(D_DIM / 256, S_DIM / 128, B_DIM);             // (4, 16, 4)
  out_gemm_kernel<<<go, thr, 0, stream>>>(wg, vtb, out);
}
